// ConvModule_2516850835886
// MI455X (gfx1250) — matrix operand order repaired and checked by reading, not compiled, not run on hardware
//
#include <hip/hip_runtime.h>
#include <hip/hip_bf16.h>

// Problem constants
#define Bsz   2
#define CIN   256
#define HH    96
#define WW    96
#define COUT  256
#define KK2   9
#define HW    (HH*WW)          // 9216
#define MROWS (Bsz*HW)         // 18432
#define KDIM  (CIN*KK2)        // 2304
#define NCOLS COUT             // 256
#define GROUPS 32
#define CPG    (COUT/GROUPS)   // 8

typedef _Float16 v16h __attribute__((ext_vector_type(16)));
typedef _Float16 v8h  __attribute__((ext_vector_type(8)));
typedef float    v8f  __attribute__((ext_vector_type(8)));
typedef int      v4i  __attribute__((ext_vector_type(4)));

// ---- async global->LDS copy support (guarded; falls back to LDS stores) ----
#if defined(__has_builtin)
# if __has_builtin(__builtin_amdgcn_global_load_async_to_lds_b128) && \
     __has_builtin(__builtin_amdgcn_s_wait_asynccnt)
#  define USE_ASYNC 1
# endif
#endif
#ifndef USE_ASYNC
# define USE_ASYNC 0
#endif

__device__ __forceinline__ void cp16_g2l(const _Float16* g, _Float16* l) {
#if USE_ASYNC
    // Signature (from toolchain diagnostic): (int4 __device__*, int4 __shared__*, imm, imm)
    __builtin_amdgcn_global_load_async_to_lds_b128(
        (__attribute__((address_space(1))) v4i*)g,
        (__attribute__((address_space(3))) v4i*)l, 0, 0);
#else
    *(v8h*)l = *(const v8h*)g;
#endif
}

__device__ __forceinline__ void async_fence() {
#if USE_ASYNC
    __builtin_amdgcn_s_wait_asynccnt(0);
#endif
}

// ---------------- Kernel 0: NCHW -> NHWC ----------------
__global__ void to_nhwc_kernel(const float* __restrict__ x, float* __restrict__ xn) {
    int bhw = blockIdx.x;           // 0..MROWS-1
    int c   = threadIdx.x;          // 0..255
    int b   = bhw / HW;
    int hw  = bhw % HW;
    xn[(size_t)bhw * CIN + c] = x[((size_t)b * CIN + c) * HW + hw];
}

// ---------------- Kernel 1: offset conv (18 ch, 3x3, pad 1) ----------------
__global__ void offset_conv_kernel(const float* __restrict__ xn,
                                   const float* __restrict__ ow,
                                   const float* __restrict__ ob,
                                   float* __restrict__ off) {
    int tid = blockIdx.x * 256 + threadIdx.x;   // (b*18 + o)*HW + hw
    int hw  = tid % HW;
    int bo  = tid / HW;
    int o   = bo % 18;
    int b   = bo / 18;
    int h = hw / WW, w = hw % WW;
    float acc = ob[o];
    for (int ky = 0; ky < 3; ++ky) {
        int ih = h - 1 + ky;
        if (ih < 0 || ih >= HH) continue;
        for (int kx = 0; kx < 3; ++kx) {
            int iw = w - 1 + kx;
            if (iw < 0 || iw >= WW) continue;
            const float* xp = &xn[((size_t)(b * HH + ih) * WW + iw) * CIN];
            const float* wp = &ow[(size_t)o * CIN * 9 + ky * 3 + kx];
            float s = 0.f;
            for (int c = 0; c < CIN; ++c) s += xp[c] * wp[(size_t)c * 9];
            acc += s;
        }
    }
    off[tid] = acc;
}

// ---------------- Kernel 2: bilinear gather -> f16 patch matrix A[M][KDIM] ----------------
__global__ void gather_kernel(const float* __restrict__ xn,
                              const float* __restrict__ off,
                              _Float16* __restrict__ Amat) {
    int bhw = blockIdx.x;           // 0..MROWS-1
    int c   = threadIdx.x;          // channel 0..255
    int b   = bhw / HW;
    int hw  = bhw % HW;
    int h = hw / WW, w = hw % WW;
    size_t arow = (size_t)bhw * KDIM;
    for (int kk = 0; kk < KK2; ++kk) {
        int ky = kk / 3, kx = kk % 3;
        float dy = off[((size_t)(b * 18 + kk * 2 + 0)) * HW + hw];
        float dx = off[((size_t)(b * 18 + kk * 2 + 1)) * HW + hw];
        float sy = (float)(h - 1 + ky) + dy;
        float sx = (float)(w - 1 + kx) + dx;
        float y0f = floorf(sy), x0f = floorf(sx);
        float fy = sy - y0f, fx = sx - x0f;
        int y0 = (int)y0f, x0 = (int)x0f;
        float acc = 0.f;
        #pragma unroll
        for (int dyc = 0; dyc < 2; ++dyc) {
            int yc = y0 + dyc;
            if (yc < 0 || yc >= HH) continue;
            float wy = dyc ? fy : (1.f - fy);
            #pragma unroll
            for (int dxc = 0; dxc < 2; ++dxc) {
                int xc = x0 + dxc;
                if (xc < 0 || xc >= WW) continue;
                float wx = dxc ? fx : (1.f - fx);
                acc += wy * wx * xn[((size_t)(b * HH + yc) * WW + xc) * CIN + c];
            }
        }
        Amat[arow + kk * CIN + c] = (_Float16)acc;
    }
}

// ---------------- Kernel 3: pack conv_w -> f16 B^T[NCOLS][KDIM], k = kk*256 + c ----------------
__global__ void pack_w_kernel(const float* __restrict__ cw, _Float16* __restrict__ BmT) {
    int tid  = blockIdx.x * 256 + threadIdx.x;  // n*KDIM + kidx  (kidx fastest)
    int kidx = tid % KDIM;
    int n    = tid / KDIM;
    int kk   = kidx >> 8;
    int c    = kidx & 255;
    BmT[tid] = (_Float16)cw[((size_t)n * CIN + c) * 9 + kk];
}

// ---------------- Kernel 4: zero GN accumulators ----------------
__global__ void zero_stats_kernel(float* stats) {
    if (threadIdx.x < Bsz * GROUPS * 2) stats[threadIdx.x] = 0.f;
}

// ---------------- Kernel 5: WMMA GEMM, async double-buffered LDS staging ----------------
#define BM 128
#define BN 64
#define BK 32
#define AST 40   // LDS row stride in halves (80B, multiple of 16B)

__global__ void gemm_wmma_kernel(const _Float16* __restrict__ A,
                                 const _Float16* __restrict__ BmT,
                                 float* __restrict__ C,
                                 float* __restrict__ stats) {
    __shared__ _Float16 As [2][BM * AST];   // A tile: 128 rows x 32 k  (M-major)
    __shared__ _Float16 BsT[2][BN * AST];   // B tile: 64 cols x 32 k  (N-major)

    const int tid   = threadIdx.x;        // 256 threads = 8 waves (wave32)
    const int lane  = tid & 31;
    const int wave  = tid >> 5;
    const int mBase = blockIdx.y * BM;    // 144 tiles
    const int nBase = blockIdx.x * BN;    // 4 tiles
    const int waveM = (wave & 3) * 32;    // 4 waves over M
    const int waveN = (wave >> 2) * 32;   // 2 waves over N

    v8f acc[2][2] = {};

    // Staging assignments (16B chunks, aligned on both sides)
    const int rowA = tid >> 1;            // 0..127
    const int segA = (tid & 1) * 16;      // 0 / 16 halves
    const int colT = tid >> 2;            // 0..63
    const int segB = (tid & 3) * 8;       // 0/8/16/24 halves

    const _Float16* gA = &A  [(size_t)(mBase + rowA) * KDIM + segA];
    const _Float16* gB = &BmT[(size_t)(nBase + colT) * KDIM + segB];

    const int hsel = (lane >> 4) * 8;     // half-wave K-select (ISA 7.12.2)
    const int lrow = lane & 15;

    // Prologue: stage tile 0 into buffer 0 (A: two 16B chunks/thread, B: one)
    cp16_g2l(gA + 0, &As [0][rowA * AST + segA]);
    cp16_g2l(gA + 8, &As [0][rowA * AST + segA + 8]);
    cp16_g2l(gB + 0, &BsT[0][colT * AST + segB]);
    async_fence();
    __syncthreads();

    int buf = 0;
    for (int k0 = 0; k0 < KDIM; k0 += BK) {
        const int nxt = buf ^ 1;
        if (k0 + BK < KDIM) {
            // Stage next tile asynchronously while computing on current one
            cp16_g2l(gA + (k0 + BK), &As [nxt][rowA * AST + segA]);
            cp16_g2l(gA + (k0 + BK + 8), &As [nxt][rowA * AST + segA + 8]);
            cp16_g2l(gB + (k0 + BK), &BsT[nxt][colT * AST + segB]);
            if (k0 + 2 * BK < KDIM)
                __builtin_prefetch(gA + k0 + 2 * BK, 0, 0);
        }

        // Fragments: two 16B LDS vector loads each, merged per ISA VGPR layout
        v16h af[2], bf[2];
        #pragma unroll
        for (int i = 0; i < 2; ++i) {
            const _Float16* base = &As[buf][(waveM + i * 16 + lrow) * AST];
            v8h lo = *(const v8h*)(base + hsel);
            v8h hi = *(const v8h*)(base + 16 + hsel);
            af[i] = __builtin_shufflevector(lo, hi, 0,1,2,3,4,5,6,7,8,9,10,11,12,13,14,15);
        }
        #pragma unroll
        for (int i = 0; i < 2; ++i) {
            const _Float16* base = &BsT[buf][(waveN + i * 16 + lrow) * AST];
            v8h lo = *(const v8h*)(base + hsel);
            v8h hi = *(const v8h*)(base + 16 + hsel);
            bf[i] = __builtin_shufflevector(lo, hi, 0,1,2,3,4,5,6,7,8,9,10,11,12,13,14,15);
        }

        #pragma unroll
        for (int i = 0; i < 2; ++i)
            #pragma unroll
            for (int j = 0; j < 2; ++j)
                acc[i][j] = __builtin_amdgcn_wmma_f32_16x16x32_f16(
                    false, af[i], false, bf[j], (short)0, acc[i][j], false, false);

        async_fence();     // this wave's async copies landed in LDS
        __syncthreads();   // all waves' copies visible; done reading 'buf'
        buf = nxt;
    }

    // Epilogue: store C + per-(batch,group) sum/sumsq for GroupNorm.
    // All 128 M-rows of a block lie in one batch image (HW = 9216 = 72*128).
    const int b    = mBase / HW;
    const int rAdd = (lane >> 4) * 8;     // D layout: VGPR v -> M=v (lanes<16) / M=8+v
    #pragma unroll
    for (int j = 0; j < 2; ++j) {
        int col = nBase + waveN + j * 16 + lrow;   // output channel
        float s = 0.f, sq = 0.f;
        #pragma unroll
        for (int i = 0; i < 2; ++i) {
            int rbase = mBase + waveM + i * 16 + rAdd;
            #pragma unroll
            for (int v = 0; v < 8; ++v) {
                float val = acc[i][j][v];
                C[(size_t)(rbase + v) * NCOLS + col] = val;
                s  += val;
                sq += val * val;
            }
        }
        int g = col >> 3;   // channel / CPG
        atomicAdd(&stats[(b * GROUPS + g) * 2 + 0], s);
        atomicAdd(&stats[(b * GROUPS + g) * 2 + 1], sq);
    }
}

// ---------------- Kernel 6: GroupNorm finalize + affine + ReLU -> NCHW ----------------
__global__ void finalize_kernel(const float* __restrict__ Cmat,
                                const float* __restrict__ stats,
                                const float* __restrict__ gamma,
                                const float* __restrict__ beta,
                                float* __restrict__ out) {
    int tid  = blockIdx.x * 256 + threadIdx.x;  // ((b*COUT + co)*HW + hw)
    int hw   = tid % HW;
    int bc   = tid / HW;
    int co   = bc & 255;
    int b    = bc >> 8;
    float val = Cmat[((size_t)(b * HW + hw)) * NCOLS + co];
    int g = co >> 3;
    const float cnt = (float)(CPG * HW);
    float s   = stats[(b * GROUPS + g) * 2 + 0];
    float sq  = stats[(b * GROUPS + g) * 2 + 1];
    float mu  = s / cnt;
    float var = sq / cnt - mu * mu;
    float r   = rsqrtf(var + 1e-5f);
    float y   = (val - mu) * r * gamma[co] + beta[co];
    out[tid]  = y > 0.f ? y : 0.f;
}

extern "C" void kernel_launch(void* const* d_in, const int* in_sizes, int n_in,
                              void* d_out, int out_size, void* d_ws, size_t ws_size,
                              hipStream_t stream) {
    const float* x     = (const float*)d_in[0];
    const float* ow    = (const float*)d_in[1];
    const float* ob    = (const float*)d_in[2];
    const float* cw    = (const float*)d_in[3];
    const float* gamma = (const float*)d_in[4];
    const float* beta  = (const float*)d_in[5];
    float* out = (float*)d_out;

    char* ws = (char*)d_ws;
    size_t off_b = 0;
    auto take = [&](size_t bytes) -> char* {
        char* p = ws + off_b;
        off_b = (off_b + bytes + 255) & ~(size_t)255;
        return p;
    };
    float*    xn    = (float*)   take((size_t)MROWS * CIN * 4);     // 18.9 MB NHWC
    float*    offp  = (float*)   take((size_t)Bsz * 18 * HW * 4);   // 1.3 MB offsets
    _Float16* Amat  = (_Float16*)take((size_t)MROWS * KDIM * 2);    // 85 MB patches (f16)
    _Float16* BmatT = (_Float16*)take((size_t)KDIM * NCOLS * 2);    // 1.2 MB weights (f16, N-major)
    float*    Cmat  = (float*)   take((size_t)MROWS * NCOLS * 4);   // 18.9 MB conv out
    float*    stats = (float*)   take(512);                          // GN sums

    to_nhwc_kernel     <<<MROWS, 256, 0, stream>>>(x, xn);
    offset_conv_kernel <<<(Bsz * 18 * HW) / 256, 256, 0, stream>>>(xn, ow, ob, offp);
    gather_kernel      <<<MROWS, 256, 0, stream>>>(xn, offp, Amat);
    pack_w_kernel      <<<KDIM, 256, 0, stream>>>(cw, BmatT);
    zero_stats_kernel  <<<1, 128, 0, stream>>>(stats);
    dim3 gg(NCOLS / BN, MROWS / BM);   // (4, 144)
    gemm_wmma_kernel   <<<gg, 256, 0, stream>>>(Amat, BmatT, Cmat, stats);
    finalize_kernel    <<<MROWS, 256, 0, stream>>>(Cmat, stats, gamma, beta, out);
}
